// LogAttention1d_3667902070985
// MI455X (gfx1250) — compile-verified, matrix operand order checked by reading, not run on hardware
//
#include <hip/hip_runtime.h>
#include <cstdint>

// ---------------------------------------------------------------------------
// MI455X (gfx1250) implementation of LogAttention1d.
//  - 1x1 convs as f16 WMMA GEMMs (v_wmma_f32_16x16x32_f16), fragment-linear
//    operand layout in workspace so fragments load as contiguous b128s.
//    Wave tile 64x64 = 16 WMMAs per K-step -> 32 FLOP/B from L2, 16
//    independent accumulation chains to cover WMMA->VALU hazard slots.
//  - log-attention levels fused per digit-window in LDS (4 passes per block
//    instead of 2*n streaming passes) -> HBM-roofline driven design.
// ---------------------------------------------------------------------------

typedef __attribute__((ext_vector_type(16))) _Float16 v16h;
typedef __attribute__((ext_vector_type(8)))  float    v8f;

#define BATCH  4
#define CCH    1024
#define NHEADS 8
#define HDIM   128
#define KDIM   64
#define WLEN   4096
#define ATT_SCALE 0.125f   // 64^-0.5

// k index inside a 16x32 f16 A fragment for (lane, element e) per ISA 7.12.2
__device__ __forceinline__ int frag_kidx(int lane, int e) {
    return ((lane >> 4) << 3) + ((e >> 3) << 4) + (((e >> 1) & 3) << 1) + (e & 1);
}

// ---------------------------------------------------------------------------
// Weight fp32 (M x K row-major) -> f16 fragment-linear: [mo][ko][lane][e]
// ---------------------------------------------------------------------------
__global__ __launch_bounds__(256) void convert_w_frag(
    const float* __restrict__ W, _Float16* __restrict__ Wf, int M, int K)
{
    const int KO = K >> 5;
    const size_t total = (size_t)M * (size_t)K;
    for (size_t idx = (size_t)blockIdx.x * 256u + threadIdx.x; idx < total;
         idx += (size_t)gridDim.x * 256u) {
        int e    = (int)(idx & 15);
        int lane = (int)((idx >> 4) & 31);
        size_t t = idx >> 9;
        int ko = (int)(t % (size_t)KO);
        int mo = (int)(t / (size_t)KO);
        int m = (mo << 4) + (lane & 15);
        int k = (ko << 5) + frag_kidx(lane, e);
        Wf[idx] = (_Float16)W[(size_t)m * K + k];
    }
}

// ---------------------------------------------------------------------------
// Activation fp32 [B][K][srcL] -> f16 B-fragment-linear: [b][no][ko][lane][e]
// (B-operand layout assumed to mirror A with M->N; zero-pad n >= nvalid)
// ---------------------------------------------------------------------------
__global__ __launch_bounds__(256) void convert_x_frag(
    const float* __restrict__ X, _Float16* __restrict__ Xf,
    int K, int Ngemm, int nvalid, int srcL)
{
    const int KO = K >> 5;
    const int NO = Ngemm >> 4;
    const size_t total = (size_t)BATCH * NO * KO * 512;
    for (size_t idx = (size_t)blockIdx.x * 256u + threadIdx.x; idx < total;
         idx += (size_t)gridDim.x * 256u) {
        int e    = (int)(idx & 15);
        int lane = (int)((idx >> 4) & 31);
        size_t t = idx >> 9;
        int ko = (int)(t % (size_t)KO); t /= (size_t)KO;
        int no = (int)(t % (size_t)NO);
        int b  = (int)(t / (size_t)NO);
        int n = (no << 4) + (lane & 15);
        int k = (ko << 5) + frag_kidx(lane, e);
        float v = (n < nvalid) ? X[((size_t)b * K + k) * srcL + n] : 0.f;
        Xf[idx] = (_Float16)v;
    }
}

// ---------------------------------------------------------------------------
// WMMA GEMM: out[b][m][n<Lstore] = (sum_k Af[m,k]*Bf[b,k,n]) * scale[m]+bias[m]
// 256 threads = 8 waves (2 M x 4 N); wave tile 64x64 = 4x4 WMMA 16x16x32.
// Block tile 128(M) x 256(N). Fragments load straight from L2-resident
// fragment-linear buffers (weights 2MB each -> fully L2-resident).
// ---------------------------------------------------------------------------
__global__ __launch_bounds__(256) void gemm_wmma_kernel(
    const _Float16* __restrict__ Af,   // [MO][KO][32][16]
    const _Float16* __restrict__ Bf,   // [B][NO][KO][32][16]
    const float* __restrict__ scale, const float* __restrict__ bias,
    float* __restrict__ out, int M, int K, int Ngemm, int Lstore)
{
    const int lane  = threadIdx.x & 31;
    const int wave  = threadIdx.x >> 5;
    const int waveM = wave >> 2;                 // 0..1
    const int waveN = wave & 3;                  // 0..3
    const int m0 = blockIdx.y * 128 + waveM * 64;
    const int n0 = blockIdx.x * 256 + waveN * 64;
    const int b  = blockIdx.z;
    const int KO = K >> 5;
    const int NO = Ngemm >> 4;

    v8f acc[4][4];
#pragma unroll
    for (int i = 0; i < 4; ++i)
#pragma unroll
        for (int j = 0; j < 4; ++j)
            acc[i][j] = (v8f){0.f,0.f,0.f,0.f,0.f,0.f,0.f,0.f};

    const _Float16* aBase = Af + (size_t)(m0 >> 4) * KO * 512;
    const _Float16* bBase = Bf + ((size_t)b * NO + (size_t)(n0 >> 4)) * KO * 512;

    for (int ko = 0; ko < KO; ++ko) {
        v16h a[4], bm[4];
#pragma unroll
        for (int i = 0; i < 4; ++i)
            a[i] = *(const v16h*)(aBase + ((size_t)i * KO + ko) * 512 + lane * 16);
#pragma unroll
        for (int j = 0; j < 4; ++j)
            bm[j] = *(const v16h*)(bBase + ((size_t)j * KO + ko) * 512 + lane * 16);
#pragma unroll
        for (int i = 0; i < 4; ++i)
#pragma unroll
            for (int j = 0; j < 4; ++j)
                acc[i][j] = __builtin_amdgcn_wmma_f32_16x16x32_f16(
                    false, a[i], false, bm[j], (short)0, acc[i][j], false, false);
    }

    // C/D layout: lanes 0-15 n=lane, rows m0..m0+7 per VGPR r; lanes 16-31 +8
#pragma unroll
    for (int i = 0; i < 4; ++i) {
        int mrow0 = m0 + i * 16 + (lane >> 4) * 8;
#pragma unroll
        for (int r = 0; r < 8; ++r) {
            int m = mrow0 + r;
            float s  = scale[m];
            float bi = bias[m];
#pragma unroll
            for (int j = 0; j < 4; ++j) {
                int n = n0 + j * 16 + (lane & 15);
                if (n < Lstore)
                    out[((size_t)b * M + m) * Lstore + n] = acc[i][j][r] * s + bi;
            }
        }
    }
}

// ---------------------------------------------------------------------------
// Fused log-attention levels. One WG owns a digit-window chunk of P=BASE^mlev
// positions (global stride S0) per (batch, head); runs mlev levels in LDS.
// dirHighFirst=0: local digits 0..mlev-1 (v1 order); =1: mlev-1..0 (v2 order).
// ---------------------------------------------------------------------------
template<int BASE>
__global__ __launch_bounds__(256) void attn_fused_kernel(
    const float* __restrict__ qk,   // [B][H*2*64][L]
    const float* __restrict__ vin,  // [B][H*128][L]
    float* __restrict__ vout,
    int L, int S0, int P, int mlev, int dirHighFirst)
{
    extern __shared__ float lds[];
    float* qs = lds;                 // [P][68]
    float* ks = qs + (size_t)P * 68; // [P][68]
    float* vs = ks + (size_t)P * 68; // [P][129]
    float* at = vs + (size_t)P * 129;// [P/BASE][BASE][BASE]

    const int tid   = threadIdx.x;
    const int chunk = blockIdx.x;    // 0 .. L/P-1
    const int h     = blockIdx.y;
    const int b     = blockIdx.z;
    const int hi = chunk / S0;
    const int lo = chunk - hi * S0;
    const size_t off = (size_t)hi * S0 * P + lo;

    const float* qg  = qk  + (((size_t)b * NHEADS + h) * 2 + 0) * KDIM * L;
    const float* kg  = qk  + (((size_t)b * NHEADS + h) * 2 + 1) * KDIM * L;
    const float* vgI = vin + ((size_t)b * NHEADS + h) * HDIM * L;
    float*       vgO = vout + ((size_t)b * NHEADS + h) * HDIM * L;

    for (int idx = tid; idx < KDIM * P; idx += 256) {
        int d = idx / P, u = idx - d * P;
        size_t ga = (size_t)d * L + off + (size_t)S0 * u;
        qs[u * 68 + d] = qg[ga];
        ks[u * 68 + d] = kg[ga];
    }
    for (int idx = tid; idx < HDIM * P; idx += 256) {
        int d = idx / P, u = idx - d * P;
        vs[u * 129 + d] = vgI[(size_t)d * L + off + (size_t)S0 * u];
    }
    __syncthreads();

    const int groups = P / BASE;
    for (int sl = 0; sl < mlev; ++sl) {
        int s = dirHighFirst ? (mlev - 1 - sl) : sl;
        int st = 1;
        for (int t = 0; t < s; ++t) st *= BASE;   // BASE^s

        // ---- scores + softmax: one thread per (group, J) ----
        if (tid < P) {
            int g = tid / BASE, J = tid - g * BASE;
            int a = g / st, c = g - a * st;
            int pbase = a * st * BASE + c;
            int posJ = pbase + J * st;
            float sc[BASE];
            float mx = -3.4e38f;
#pragma unroll
            for (int j = 0; j < BASE; ++j) {
                int posj = pbase + j * st;
                const float4* qp = (const float4*)(qs + (size_t)posj * 68);
                const float4* kp = (const float4*)(ks + (size_t)posJ * 68);
                float dot = 0.f;
#pragma unroll
                for (int d4 = 0; d4 < 16; ++d4) {
                    float4 qv = qp[d4], kv = kp[d4];
                    dot += qv.x * kv.x + qv.y * kv.y + qv.z * kv.z + qv.w * kv.w;
                }
                sc[j] = dot * ATT_SCALE;
                mx = fmaxf(mx, sc[j]);
            }
            float sum = 0.f;
#pragma unroll
            for (int j = 0; j < BASE; ++j) { sc[j] = __expf(sc[j] - mx); sum += sc[j]; }
            float inv = 1.f / sum;
#pragma unroll
            for (int j = 0; j < BASE; ++j) at[(g * BASE + J) * BASE + j] = sc[j] * inv;
        }
        __syncthreads();

        // ---- apply: each thread owns (group, d) -> no cross-thread hazard ----
        for (int idx = tid; idx < groups * HDIM; idx += 256) {
            int g = idx >> 7, d = idx & 127;
            int a = g / st, c = g - a * st;
            int pbase = a * st * BASE + c;
            float vold[BASE], vnew[BASE];
#pragma unroll
            for (int j = 0; j < BASE; ++j) vold[j] = vs[(pbase + j * st) * 129 + d];
#pragma unroll
            for (int J = 0; J < BASE; ++J) {
                float acc = 0.f;
#pragma unroll
                for (int j = 0; j < BASE; ++j) acc += at[(g * BASE + J) * BASE + j] * vold[j];
                vnew[J] = acc;
            }
#pragma unroll
            for (int J = 0; J < BASE; ++J) vs[(pbase + J * st) * 129 + d] = vnew[J];
        }
        __syncthreads();
    }

    for (int idx = tid; idx < HDIM * P; idx += 256) {
        int d = idx / P, u = idx - d * P;
        vgO[(size_t)d * L + off + (size_t)S0 * u] = vs[u * 129 + d];
    }
}

// ---------------------------------------------------------------------------
// y[b,c,l<w] = xin[b,c,l] + va[b,c,l]   (va has row stride Lv)
// ---------------------------------------------------------------------------
__global__ __launch_bounds__(256) void add_v_kernel(
    float* __restrict__ y, const float* __restrict__ xin,
    const float* __restrict__ va, int w, int Lv)
{
    const size_t total = (size_t)BATCH * CCH * w;
    for (size_t idx = (size_t)blockIdx.x * 256u + threadIdx.x; idx < total;
         idx += (size_t)gridDim.x * 256u) {
        int l = (int)(idx % (size_t)w);
        size_t bc = idx / (size_t)w;
        y[idx] = xin[idx] + va[bc * Lv + l];
    }
}

// ---------------------------------------------------------------------------
// y += va + dwconv3(v0)*spe+bpe  (on the L-domain, truncated to w)
// ---------------------------------------------------------------------------
__global__ __launch_bounds__(256) void add_v_pe_kernel(
    float* __restrict__ y, const float* __restrict__ va,
    const float* __restrict__ v0, const float* __restrict__ wpe,
    const float* __restrict__ spe, const float* __restrict__ bpe,
    int w, int Lv)
{
    const size_t total = (size_t)BATCH * CCH * w;
    for (size_t idx = (size_t)blockIdx.x * 256u + threadIdx.x; idx < total;
         idx += (size_t)gridDim.x * 256u) {
        int l = (int)(idx % (size_t)w);
        size_t bc = idx / (size_t)w;
        int c = (int)(bc % CCH);
        const float* r = v0 + bc * Lv;
        float lf = (l > 0)      ? r[l - 1] : 0.f;
        float md = r[l];
        float rt = (l + 1 < Lv) ? r[l + 1] : 0.f;
        float pe = (wpe[c * 3 + 0] * lf + wpe[c * 3 + 1] * md + wpe[c * 3 + 2] * rt)
                   * spe[c] + bpe[c];
        y[idx] += va[bc * Lv + l] + pe;
    }
}

// ---------------------------------------------------------------------------
extern "C" void kernel_launch(void* const* d_in, const int* in_sizes, int n_in,
                              void* d_out, int out_size, void* d_ws, size_t ws_size,
                              hipStream_t stream) {
    (void)in_sizes; (void)n_in; (void)out_size; (void)ws_size;
    const float* x     = (const float*)d_in[0];
    const float* wqk2  = (const float*)d_in[1];
    const float* sqk2  = (const float*)d_in[2];
    const float* bqk2  = (const float*)d_in[3];
    const float* wv2   = (const float*)d_in[4];
    const float* sv2   = (const float*)d_in[5];
    const float* bv2   = (const float*)d_in[6];
    const float* wpe2  = (const float*)d_in[7];
    const float* spe2  = (const float*)d_in[8];
    const float* bpe2  = (const float*)d_in[9];
    const float* wqk3  = (const float*)d_in[10];
    const float* sqk3  = (const float*)d_in[11];
    const float* bqk3  = (const float*)d_in[12];
    const float* wv3   = (const float*)d_in[13];
    const float* sv3   = (const float*)d_in[14];
    const float* bv3   = (const float*)d_in[15];
    const float* wpe3  = (const float*)d_in[16];
    const float* spe3  = (const float*)d_in[17];
    const float* bpe3  = (const float*)d_in[18];
    const float* wproj = (const float*)d_in[19];
    const float* sproj = (const float*)d_in[20];
    const float* bproj = (const float*)d_in[21];
    float* y = (float*)d_out;          // y accumulator doubles as final output

    // workspace carve-up
    char* ws = (char*)d_ws;
    size_t off = 0;
    auto take = [&](size_t bytes) -> char* {
        char* p = ws + off;
        off += (bytes + 255) & ~(size_t)255;
        return p;
    };
    const size_t WBYTES = 2ull * CCH * CCH;            // 2 MB per f16 weight
    _Float16* WfQK2 = (_Float16*)take(WBYTES);
    _Float16* WfV2  = (_Float16*)take(WBYTES);
    _Float16* WfQK3 = (_Float16*)take(WBYTES);
    _Float16* WfV3  = (_Float16*)take(WBYTES);
    _Float16* WfPRJ = (_Float16*)take(WBYTES);
    const int L2 = 4096, L3 = 6561;
    const int NG2 = 4096, NG3 = 6656;                  // GEMM-N (multiple of 256)
    _Float16* Xf = (_Float16*)take(2ull * BATCH * CCH * NG3);
    float* qkb = (float*)take(4ull * BATCH * CCH * L3);
    float* v0b = (float*)take(4ull * BATCH * CCH * L3);
    float* vab = (float*)take(4ull * BATCH * CCH * L3);

    const dim3 blk(256);
    const dim3 cgrid(4096);
    const size_t sh2 = (size_t)(2 * 64 * 68 + 64 * 129 + 64 * 2) * 4;   // 68352 B
    const size_t sh3 = (size_t)(2 * 81 * 68 + 81 * 129 + 81 * 3) * 4;   // 86832 B

    // weights -> f16 fragment layout (L2-resident for all GEMMs)
    convert_w_frag<<<cgrid, blk, 0, stream>>>(wqk2,  WfQK2, CCH, CCH);
    convert_w_frag<<<cgrid, blk, 0, stream>>>(wv2,   WfV2,  CCH, CCH);
    convert_w_frag<<<cgrid, blk, 0, stream>>>(wqk3,  WfQK3, CCH, CCH);
    convert_w_frag<<<cgrid, blk, 0, stream>>>(wv3,   WfV3,  CCH, CCH);
    convert_w_frag<<<cgrid, blk, 0, stream>>>(wproj, WfPRJ, CCH, CCH);

    // ================= block 2: base=2, n=12, L=4096 =================
    convert_x_frag<<<cgrid, blk, 0, stream>>>(x, Xf, CCH, NG2, WLEN, WLEN);
    gemm_wmma_kernel<<<dim3(NG2/256, CCH/128, BATCH), blk, 0, stream>>>(
        WfQK2, Xf, sqk2, bqk2, qkb, CCH, CCH, NG2, L2);
    gemm_wmma_kernel<<<dim3(NG2/256, CCH/128, BATCH), blk, 0, stream>>>(
        WfV2, Xf, sv2, bv2, v0b, CCH, CCH, NG2, L2);
    // v1: digits 0..5 then 6..11 (low digit first)
    attn_fused_kernel<2><<<dim3(64, NHEADS, BATCH), blk, sh2, stream>>>(
        qkb, v0b, vab, L2, 1, 64, 6, 0);
    attn_fused_kernel<2><<<dim3(64, NHEADS, BATCH), blk, sh2, stream>>>(
        qkb, vab, vab, L2, 64, 64, 6, 0);
    add_v_kernel<<<cgrid, blk, 0, stream>>>(y, x, vab, WLEN, L2);
    // v2: digits 11..6 then 5..0 (high digit first)
    attn_fused_kernel<2><<<dim3(64, NHEADS, BATCH), blk, sh2, stream>>>(
        qkb, v0b, vab, L2, 64, 64, 6, 1);
    attn_fused_kernel<2><<<dim3(64, NHEADS, BATCH), blk, sh2, stream>>>(
        qkb, vab, vab, L2, 1, 64, 6, 1);
    add_v_pe_kernel<<<cgrid, blk, 0, stream>>>(y, vab, v0b, wpe2, spe2, bpe2, WLEN, L2);

    // ================= block 3: base=3, n=8, L=6561 =================
    convert_x_frag<<<cgrid, blk, 0, stream>>>(y, Xf, CCH, NG3, WLEN, WLEN);
    gemm_wmma_kernel<<<dim3(NG3/256, CCH/128, BATCH), blk, 0, stream>>>(
        WfQK3, Xf, sqk3, bqk3, qkb, CCH, CCH, NG3, L3);
    gemm_wmma_kernel<<<dim3(NG3/256, CCH/128, BATCH), blk, 0, stream>>>(
        WfV3, Xf, sv3, bv3, v0b, CCH, CCH, NG3, L3);
    attn_fused_kernel<3><<<dim3(81, NHEADS, BATCH), blk, sh3, stream>>>(
        qkb, v0b, vab, L3, 1, 81, 4, 0);
    attn_fused_kernel<3><<<dim3(81, NHEADS, BATCH), blk, sh3, stream>>>(
        qkb, vab, vab, L3, 81, 81, 4, 0);
    add_v_kernel<<<cgrid, blk, 0, stream>>>(y, y, vab, WLEN, L3);
    attn_fused_kernel<3><<<dim3(81, NHEADS, BATCH), blk, sh3, stream>>>(
        qkb, v0b, vab, L3, 81, 81, 4, 1);
    attn_fused_kernel<3><<<dim3(81, NHEADS, BATCH), blk, sh3, stream>>>(
        qkb, vab, vab, L3, 1, 81, 4, 1);
    add_v_pe_kernel<<<cgrid, blk, 0, stream>>>(y, vab, v0b, wpe3, spe3, bpe3, WLEN, L3);

    // ================= final projection =================
    convert_x_frag<<<cgrid, blk, 0, stream>>>(y, Xf, CCH, NG2, WLEN, WLEN);
    gemm_wmma_kernel<<<dim3(NG2/256, CCH/128, BATCH), blk, 0, stream>>>(
        WfPRJ, Xf, sproj, bproj, y, CCH, CCH, NG2, WLEN);
}
